// TriangularUpdate_52810917871757
// MI455X (gfx1250) — matrix operand order repaired and checked by reading, not compiled, not run on hardware
//
#include <hip/hip_runtime.h>
#include <hip/hip_bf16.h>
#include <stdint.h>

typedef _Float16 h16v __attribute__((ext_vector_type(16)));
typedef _Float16 h8v  __attribute__((ext_vector_type(8)));
typedef float    f8v  __attribute__((ext_vector_type(8)));

union H16 { h16v v; h8v h[2]; _Float16 e[16]; };

#define PLANE (512 * 512)

__device__ __forceinline__ unsigned lds_lo(const void* p) {
  // low 32 bits of a generic pointer to LDS == wave-relative LDS byte offset
  return (unsigned)(unsigned long long)p;
}
__device__ __forceinline__ float sigmoidf(float x) { return 1.0f / (1.0f + __expf(-x)); }

__device__ __forceinline__ f8v wmma_f16(const H16& a, const H16& b, f8v c) {
  return __builtin_amdgcn_wmma_f32_16x16x32_f16(false, a.v, false, b.v, (short)0, c, false, false);
}

// ---------------------------------------------------------------------------
// Kernel 1: convert 6 weight matrices [k][n] f32 -> transposed f16 [n][k]
// order: le, re, lu, ru, fu, rs
// ---------------------------------------------------------------------------
__global__ __launch_bounds__(256) void k_cvt_w(
    const float* __restrict__ w0, const float* __restrict__ w1,
    const float* __restrict__ w2, const float* __restrict__ w3,
    const float* __restrict__ w4, const float* __restrict__ w5,
    _Float16* __restrict__ Wt) {
  int idx = blockIdx.x * 256 + threadIdx.x;  // 0 .. 6*16384-1
  int m = idx >> 14, rem = idx & 16383;
  int k = rem >> 7, n = rem & 127;
  const float* W = (m == 0) ? w0 : (m == 1) ? w1 : (m == 2) ? w2
                 : (m == 3) ? w3 : (m == 4) ? w4 : w5;
  Wt[m * 16384 + n * 128 + k] = (_Float16)W[k * 128 + n];
}

// ---------------------------------------------------------------------------
// Kernel 2: LayerNorm(d) + 5 projections + sigmoid gating, 16 rows / block.
// Outputs: lu_t[c][i][j] f16, ru_t[c][j][k] f16 (channel planes), fu_h[row][d] f16
// ---------------------------------------------------------------------------
__global__ __launch_bounds__(256) void k_ln_proj(
    const float* __restrict__ x, const float* __restrict__ mask,
    const float* __restrict__ eg, const float* __restrict__ eb,
    const float* __restrict__ le_b, const float* __restrict__ re_b,
    const float* __restrict__ lu_b, const float* __restrict__ ru_b,
    const float* __restrict__ fu_b,
    const _Float16* __restrict__ Wt,
    _Float16* __restrict__ lu_t, _Float16* __restrict__ ru_t,
    _Float16* __restrict__ fu_h) {
  __shared__ __align__(16) _Float16 xh[16][128];
  __shared__ float mrow[16];
  __shared__ __align__(16) _Float16 st[8][16][16];

  const int t  = threadIdx.x;
  const int r0 = blockIdx.x * 16;   // flattened (i,j) row base; 16-row tiles never straddle i
  const int i  = r0 >> 9;
  const int j0 = r0 & 511;

  // ---- LayerNorm over d=128 for 16 rows (16 lanes per row, 8 elems/lane) ----
  {
    int row = t >> 4;
    int c0  = (t & 15) * 8;
    const float* xp = x + (size_t)(r0 + row) * 128 + c0;
    float v[8];
#pragma unroll
    for (int q = 0; q < 8; ++q) v[q] = xp[q];
    float s = 0.f, s2 = 0.f;
#pragma unroll
    for (int q = 0; q < 8; ++q) { s += v[q]; s2 += v[q] * v[q]; }
#pragma unroll
    for (int m = 1; m < 16; m <<= 1) {
      s  += __shfl_xor(s,  m, 32);
      s2 += __shfl_xor(s2, m, 32);
    }
    float mu   = s * (1.f / 128.f);
    float var  = s2 * (1.f / 128.f) - mu * mu;
    float rstd = rsqrtf(var + 1e-5f);
#pragma unroll
    for (int q = 0; q < 8; ++q) {
      float nx = (v[q] - mu) * rstd * eg[c0 + q] + eb[c0 + q];
      xh[row][c0 + q] = (_Float16)nx;
    }
    if (t < 16) mrow[t] = mask[(size_t)i * 512 + j0 + t];
  }
  __syncthreads();

  // ---- 5 projections via WMMA: M=16 rows, N=16 channels per wave, K=128 ----
  const int w  = t >> 5;
  const int ln = t & 31;
  const int Ml = ln & 15;
  const int hh = ln >> 4;
  const int n0 = w * 16;
  const int nn = n0 + Ml;

  H16 af[4];
#pragma unroll
  for (int ks = 0; ks < 4; ++ks) {
    int K0 = ks * 32;
    af[ks].h[0] = *(const h8v*)&xh[Ml][K0 + 16 * hh];
    af[ks].h[1] = *(const h8v*)&xh[Ml][K0 + 8 + 16 * hh];
  }

  f8v acc[5];
#pragma unroll
  for (int p = 0; p < 5; ++p) {
    f8v a = {};
#pragma unroll
    for (int ks = 0; ks < 4; ++ks) {
      H16 bf;
      bf.v = *(const h16v*)&Wt[p * 16384 + nn * 128 + ks * 32 + 16 * hh];
      a = wmma_f16(af[ks], bf, a);
    }
    acc[p] = a;
  }

  float bl = le_b[nn], br = re_b[nn], blu = lu_b[nn], bru = ru_b[nn], bfu = fu_b[nn];
  _Float16 luv[8], ruv[8], fuv[8];
#pragma unroll
  for (int v8 = 0; v8 < 8; ++v8) {
    int M = v8 + 8 * hh;
    float mm = mrow[M];
    float le = (acc[0][v8] + bl) * mm;
    float re = (acc[1][v8] + br) * mm;
    luv[v8] = (_Float16)(le * sigmoidf(acc[2][v8] + blu));
    ruv[v8] = (_Float16)(re * sigmoidf(acc[3][v8] + bru));
    fuv[v8] = (_Float16)(sigmoidf(acc[4][v8] + bfu));
  }

  // lu: stage [N][M], store rows into channel plane lu_t[c][i][j0+M]
#pragma unroll
  for (int v8 = 0; v8 < 8; ++v8) st[w][Ml][v8 + 8 * hh] = luv[v8];
  {
    h8v val = *(const h8v*)&st[w][Ml][hh * 8];
    *(h8v*)&lu_t[((size_t)(n0 + Ml) * 512 + i) * 512 + j0 + hh * 8] = val;
  }
  // ru: same plane layout ru_t[c][j][k]
#pragma unroll
  for (int v8 = 0; v8 < 8; ++v8) st[w][Ml][v8 + 8 * hh] = ruv[v8];
  {
    h8v val = *(const h8v*)&st[w][Ml][hh * 8];
    *(h8v*)&ru_t[((size_t)(n0 + Ml) * 512 + i) * 512 + j0 + hh * 8] = val;
  }
  // fu: stage [M][N], store row-major fu_h[row][d]
#pragma unroll
  for (int v8 = 0; v8 < 8; ++v8) st[w][v8 + 8 * hh][Ml] = fuv[v8];
  {
    h8v val = *(const h8v*)&st[w][Ml][hh * 8];
    *(h8v*)&fu_h[(size_t)(r0 + Ml) * 128 + n0 + hh * 8] = val;
  }
}

// ---------------------------------------------------------------------------
// Kernel 3: per-channel einsum GEMM  u[c][i][j] = sum_k lu_t[c][i][k]*ru_t[c][j][k]
// Block: 128(i) x 64(j) tile; 8 waves, each a 32x32 sub-tile (4 acc chains).
// K=512 in 16 stages, double-buffered b128 async->LDS (3 async per thread per
// stage, s_wait_asynccnt 3 to overlap the next stage's DMA with compute).
// ---------------------------------------------------------------------------
__global__ __launch_bounds__(256) void k_einsum(
    const _Float16* __restrict__ lu_t, const _Float16* __restrict__ ru_t,
    float* __restrict__ u) {
  __shared__ __align__(16) _Float16 As[2][128][32];
  __shared__ __align__(16) _Float16 Bs[2][64][32];

  const int t  = threadIdx.x;
  const int j0 = blockIdx.x * 64;
  const int i0 = blockIdx.y * 128;
  const int c  = blockIdx.z;

  const int w = t >> 5, ln = t & 31, Ml = ln & 15, hh = ln >> 4;
  const int wi = w >> 1, wj = w & 1;   // 4 x 2 waves of 32x32

  const size_t baseA = ((size_t)c * 512 + i0) * 512;
  const size_t baseB = ((size_t)c * 512 + j0) * 512;

  // A: 128 rows x 64B = 512 x 16B chunks (2 per thread); B: 64 rows = 256 chunks
  auto issue = [&](int buf, int ks) {
    const int K0 = ks * 32;
#pragma unroll
    for (int h2 = 0; h2 < 2; ++h2) {
      int idx = t + 256 * h2;
      int rA = idx >> 2, qA = idx & 3;
      uint64_t g = (uint64_t)(uintptr_t)(lu_t + baseA + (size_t)rA * 512 + K0) + (uint64_t)qA * 16;
      unsigned l = lds_lo(&As[buf][rA][qA * 8]);
      asm volatile("global_load_async_to_lds_b128 %0, %1, off" :: "v"(l), "v"(g) : "memory");
    }
    {
      int rB = t >> 2, qB = t & 3;
      uint64_t g = (uint64_t)(uintptr_t)(ru_t + baseB + (size_t)rB * 512 + K0) + (uint64_t)qB * 16;
      unsigned l = lds_lo(&Bs[buf][rB][qB * 8]);
      asm volatile("global_load_async_to_lds_b128 %0, %1, off" :: "v"(l), "v"(g) : "memory");
    }
  };

  f8v acc[4] = {f8v{}, f8v{}, f8v{}, f8v{}};
  issue(0, 0);
  for (int ks = 0; ks < 16; ++ks) {
    int buf = ks & 1;
    if (ks < 15) {
      issue(buf ^ 1, ks + 1);
      asm volatile("s_wait_asynccnt 0x3" ::: "memory");  // the 3 oldest (current stage) done
    } else {
      asm volatile("s_wait_asynccnt 0x0" ::: "memory");
    }
    __syncthreads();
    H16 a[2], b[2];
#pragma unroll
    for (int ii = 0; ii < 2; ++ii) {
      a[ii].h[0] = *(const h8v*)&As[buf][32 * wi + 16 * ii + Ml][16 * hh];
      a[ii].h[1] = *(const h8v*)&As[buf][32 * wi + 16 * ii + Ml][8 + 16 * hh];
      b[ii].v    = *(const h16v*)&Bs[buf][32 * wj + 16 * ii + Ml][16 * hh];
    }
#pragma unroll
    for (int ii = 0; ii < 2; ++ii)
#pragma unroll
      for (int jj = 0; jj < 2; ++jj)
        acc[ii * 2 + jj] = wmma_f16(a[ii], b[jj], acc[ii * 2 + jj]);
    __syncthreads();
  }

  float* up = u + (size_t)c * PLANE;
#pragma unroll
  for (int ii = 0; ii < 2; ++ii)
#pragma unroll
    for (int jj = 0; jj < 2; ++jj)
#pragma unroll
      for (int v8 = 0; v8 < 8; ++v8) {
        int ig = i0 + 32 * wi + 16 * ii + v8 + 8 * hh;
        int jg = j0 + 32 * wj + 16 * jj + Ml;
        up[(size_t)ig * 512 + jg] = acc[ii * 2 + jj][v8];
      }
}

// ---------------------------------------------------------------------------
// Kernel 4: LN over c + rs projection (WMMA) + gating -> final output f32
// Block handles (i fixed, 64 j's), all 128 channels / 128 output dims.
// ---------------------------------------------------------------------------
__global__ __launch_bounds__(256) void k_out(
    const float* __restrict__ u_, const float* __restrict__ cg, const float* __restrict__ cb,
    const float* __restrict__ rs_b, const _Float16* __restrict__ Wt,
    const _Float16* __restrict__ fu_h, const float* __restrict__ mask,
    float* __restrict__ out) {
  __shared__ __align__(16) float us[128][65];       // padded to dodge bank conflicts
  __shared__ __align__(16) _Float16 xt[64][128];
  __shared__ float ps[4][64], pq[4][64];
  __shared__ float muv[64], rsv[64];

  const int t  = threadIdx.x;
  const int j0 = blockIdx.x * 64;
  const int i  = blockIdx.y;

  // stage u[c][i][j0..j0+63]
#pragma unroll
  for (int q = 0; q < 8; ++q) {
    int idx  = t + 256 * q;  // 2048 float4 chunks
    int c    = idx >> 4;
    int col4 = idx & 15;
    float4 v = *(const float4*)(u_ + (size_t)c * PLANE + (size_t)i * 512 + j0 + col4 * 4);
    us[c][col4 * 4 + 0] = v.x; us[c][col4 * 4 + 1] = v.y;
    us[c][col4 * 4 + 2] = v.z; us[c][col4 * 4 + 3] = v.w;
  }
  __syncthreads();

  // LN statistics over c per column j
  {
    int j = t & 63, s = t >> 6;
    float sm = 0.f, sq = 0.f;
#pragma unroll 8
    for (int c = s * 32; c < s * 32 + 32; ++c) { float v = us[c][j]; sm += v; sq += v * v; }
    ps[s][j] = sm; pq[s][j] = sq;
  }
  __syncthreads();
  if (t < 64) {
    float sm = ps[0][t] + ps[1][t] + ps[2][t] + ps[3][t];
    float sq = pq[0][t] + pq[1][t] + pq[2][t] + pq[3][t];
    float mu = sm * (1.f / 128.f);
    float var = sq * (1.f / 128.f) - mu * mu;
    muv[t] = mu; rsv[t] = rsqrtf(var + 1e-5f);
  }
  __syncthreads();

  // normalized, transposed to [j][c] f16
#pragma unroll
  for (int q = 0; q < 32; ++q) {
    int idx = t + 256 * q;  // 8192 elements
    int j = idx >> 7, c = idx & 127;
    xt[j][c] = (_Float16)((us[c][j] - muv[j]) * rsv[j] * cg[c] + cb[c]);
  }
  __syncthreads();

  // rs projection: [64 j x 128 c] x [128 c x 128 d], 4 tasks per wave
  const int w = t >> 5, ln = t & 31, Ml = ln & 15, hh = ln >> 4;
  const _Float16* W5 = Wt + 5 * 16384;
#pragma unroll
  for (int tk = 0; tk < 4; ++tk) {
    int task = w * 4 + tk;
    int mt = task >> 3, nt = task & 7;
    f8v acc = {};
#pragma unroll
    for (int ks = 0; ks < 4; ++ks) {
      H16 a, b;
      a.h[0] = *(const h8v*)&xt[16 * mt + Ml][ks * 32 + 16 * hh];
      a.h[1] = *(const h8v*)&xt[16 * mt + Ml][ks * 32 + 8 + 16 * hh];
      b.v    = *(const h16v*)&W5[(16 * nt + Ml) * 128 + ks * 32 + 16 * hh];
      acc = wmma_f16(a, b, acc);
    }
    int dg = 16 * nt + Ml;
    float bias = rs_b[dg];
#pragma unroll
    for (int v8 = 0; v8 < 8; ++v8) {
      int jg = j0 + 16 * mt + v8 + 8 * hh;
      size_t ridx = (size_t)i * 512 + jg;
      float fv = (float)fu_h[ridx * 128 + dg];
      float mv = mask[ridx];
      out[ridx * 128 + dg] = (acc[v8] + bias) * fv * mv;
    }
  }
}

// ---------------------------------------------------------------------------
extern "C" void kernel_launch(void* const* d_in, const int* in_sizes, int n_in,
                              void* d_out, int out_size, void* d_ws, size_t ws_size,
                              hipStream_t stream) {
  const float* pair_rep = (const float*)d_in[0];
  const float* mask     = (const float*)d_in[1];
  const float* e_gamma  = (const float*)d_in[2];
  const float* e_beta   = (const float*)d_in[3];
  const float* le_w     = (const float*)d_in[4];
  const float* le_b     = (const float*)d_in[5];
  const float* re_w     = (const float*)d_in[6];
  const float* re_b     = (const float*)d_in[7];
  const float* lu_w     = (const float*)d_in[8];
  const float* lu_b     = (const float*)d_in[9];
  const float* ru_w     = (const float*)d_in[10];
  const float* ru_b     = (const float*)d_in[11];
  const float* fu_w     = (const float*)d_in[12];
  const float* fu_b     = (const float*)d_in[13];
  const float* c_gamma  = (const float*)d_in[14];
  const float* c_beta   = (const float*)d_in[15];
  const float* rs_w     = (const float*)d_in[16];
  const float* rs_b     = (const float*)d_in[17];
  float* out = (float*)d_out;

  // workspace layout (bytes, 256-aligned)
  char* ws = (char*)d_ws;
  _Float16* Wt   = (_Float16*)(ws);                                   // 196608 B
  _Float16* lu_t = (_Float16*)(ws + 196608);                          // 64 MiB
  _Float16* ru_t = (_Float16*)(ws + 196608 + 67108864ull);            // 64 MiB
  _Float16* fu_h = (_Float16*)(ws + 196608 + 2ull * 67108864ull);     // 64 MiB
  float*    u    = (float*)   (ws + 196608 + 3ull * 67108864ull);     // 128 MiB

  // 1) weight convert+transpose
  k_cvt_w<<<384, 256, 0, stream>>>(le_w, re_w, lu_w, ru_w, fu_w, rs_w, Wt);

  // 2) LN + projections + gating (16384 row-tiles of 16)
  k_ln_proj<<<16384, 256, 0, stream>>>(pair_rep, mask, e_gamma, e_beta,
                                       le_b, re_b, lu_b, ru_b, fu_b,
                                       Wt, lu_t, ru_t, fu_h);

  // 3) per-channel einsum GEMM (8 j-tiles x 4 i-tiles x 128 channels)
  k_einsum<<<dim3(8, 4, 128), 256, 0, stream>>>(lu_t, ru_t, u);

  // 4) LN(c) + rs projection + gating (8 j-blocks x 512 i)
  k_out<<<dim3(8, 512), 256, 0, stream>>>(u, c_gamma, c_beta, rs_b, Wt,
                                          fu_h, mask, out);
}
